// BasicGGNN_51677046505789
// MI455X (gfx1250) — compile-verified, matrix operand order checked by reading, not run on hardware
//
#include <hip/hip_runtime.h>

#define DI __device__ __forceinline__

typedef __attribute__((ext_vector_type(16))) __bf16 v16bf;
typedef __attribute__((ext_vector_type(8)))  __bf16 bf16x8;
typedef __attribute__((ext_vector_type(8)))  float  v8f;

constexpr int NN    = 50000;
constexpr int NE    = 800000;
constexpr int IND   = 128;
constexpr int H     = 256;
constexpr int NT    = 13;
constexpr int NSTEP = 6;
constexpr int NG    = 64;
constexpr int LSTR  = H + 8;             // LDS row stride (bf16 elems), keeps 16B align
constexpr int MAXTILES = NE / 64 + NT;   // upper bound on 64-edge tiles over etype groups

// ---- WMMA fragment helpers -------------------------------------------------
// 16-bit operand layout (ISA 7.12.2): lane holds two contiguous 8-elem chunks of
// its row at K offsets {base, base+16}, base = (lane>=16 ? 8 : 0).
DI v16bf frag_row(const __bf16* rowp, int k0, int lane) {
  int kb = k0 + ((lane >> 4) << 3);
  union { v16bf v; bf16x8 h[2]; } u;
  u.h[0] = *(const bf16x8*)(rowp + kb);
  u.h[1] = *(const bf16x8*)(rowp + kb + 16);
  return u.v;
}

DI v8f wmma_bf16(v16bf a, v16bf b, v8f c) {
  return __builtin_amdgcn_wmma_f32_16x16x32_bf16(false, a, false, b, (short)0, c,
                                                 false, false);
}

DI float sigm(float x) { return 1.0f / (1.0f + __expf(-x)); }

// ---- prep kernels ----------------------------------------------------------
// meta layout (64 ints): [0..12] counts, [16..29] offsets, [32..45] tileBase(64),
//                        [48..60] scatter cursors
__global__ void k_zero_meta(int* meta) { if (threadIdx.x < 64) meta[threadIdx.x] = 0; }

__global__ void k_hist(const int* __restrict__ ety, int* meta) {
  int i = blockIdx.x * 256 + threadIdx.x;
  if (i < NE) atomicAdd(&meta[ety[i]], 1);
}

__global__ void k_scan(int* meta) {
  if (blockIdx.x == 0 && threadIdx.x == 0) {
    int off = 0, tb = 0;
    for (int e = 0; e < NT; e++) {
      meta[16 + e] = off; meta[32 + e] = tb; meta[48 + e] = 0;
      off += meta[e];
      tb  += (meta[e] + 63) >> 6;
    }
    meta[16 + NT] = off; meta[32 + NT] = tb;
  }
}

__global__ void k_scatter(const int* __restrict__ ety, int* meta, int* __restrict__ perm) {
  int i = blockIdx.x * 256 + threadIdx.x;
  if (i < NE) {
    int e = ety[i];
    int pos = atomicAdd(&meta[48 + e], 1);
    perm[meta[16 + e] + pos] = i;
  }
}

// W[e][d][f] (f32) -> Wt[e][f][d] (bf16): B-fragment rows become contiguous
__global__ void k_conv_wt(const float* __restrict__ W, __bf16* __restrict__ Wt) {
  int idx = blockIdx.x * 256 + threadIdx.x;
  if (idx < NT * H * H) {
    int e = idx / (H * H); int rem = idx % (H * H);
    int d = rem / H;       int f = rem % H;
    Wt[(size_t)e * H * H + (size_t)f * H + d] = (__bf16)W[idx];
  }
}

__global__ void k_f2b(const float* __restrict__ s, __bf16* __restrict__ d, int n) {
  int i = (blockIdx.x * 256 + threadIdx.x) * 4;
  if (i < n) {
    float4 v = *(const float4*)(s + i);
    d[i] = (__bf16)v.x; d[i+1] = (__bf16)v.y; d[i+2] = (__bf16)v.z; d[i+3] = (__bf16)v.w;
  }
}

__global__ void k_zerof(float* __restrict__ p, int n) {
  int i = (blockIdx.x * 256 + threadIdx.x) * 4;
  if (i < n) { float4 z = {0.f, 0.f, 0.f, 0.f}; *(float4*)(p + i) = z; }
}

// h0 = [feat | 0], written both as f32 and bf16
__global__ void k_inith(const float* __restrict__ feat, float* __restrict__ h,
                        __bf16* __restrict__ hb) {
  int idx = blockIdx.x * 256 + threadIdx.x;
  if (idx < NN * H) {
    int n = idx >> 8, c = idx & 255;
    float v = (c < IND) ? feat[n * IND + c] : 0.0f;
    h[idx]  = v;
    hb[idx] = (__bf16)v;
  }
}

// ---- message kernel: 64 edges/WG, msg = h[src] @ W[etype], scatter to a[dst]
__global__ __launch_bounds__(128) void k_msg(
    const __bf16* __restrict__ hb, const __bf16* __restrict__ Wt,
    const int* __restrict__ perm, const int* __restrict__ src,
    const int* __restrict__ dst, const int* __restrict__ meta,
    float* __restrict__ a)
{
  __shared__ __bf16 ldsA[64 * LSTR];
  __shared__ int ldsD[64];
  int t = blockIdx.x;
  if (t >= meta[32 + NT]) return;
  int e = 0;
  while (t >= meta[32 + e + 1]) e++;
  int cnt = meta[e], off = meta[16 + e];
  int base = (t - meta[32 + e]) * 64;

  int tid = threadIdx.x;
  // gather 64 source rows (256 bf16 each) into LDS; 512 chunk-tasks of 32 bf16
  for (int i = tid; i < 512; i += 128) {
    int row = i >> 3, ch = i & 7;
    int eidx = base + row;
    int pe = (eidx < cnt) ? perm[off + eidx] : -1;
    if (ch == 0) ldsD[row] = (pe >= 0) ? dst[pe] : -1;
    __bf16* ldst = ldsA + row * LSTR + ch * 32;
    if (pe >= 0) {
      const bf16x8* sp = (const bf16x8*)(hb + (size_t)src[pe] * H + ch * 32);
      bf16x8* dp = (bf16x8*)ldst;
      dp[0] = sp[0]; dp[1] = sp[1]; dp[2] = sp[2]; dp[3] = sp[3];
    } else {
      for (int j = 0; j < 32; j++) ldst[j] = (__bf16)0.0f;
    }
  }
  __syncthreads();

  int lane = tid & 31, wave = tid >> 5;
  int rb = (lane >> 4) << 3;
  // hoist dst indices for this lane's 32 output rows into registers
  int dnreg[32];
#pragma unroll
  for (int mt = 0; mt < 4; mt++)
#pragma unroll
    for (int v = 0; v < 8; v++)
      dnreg[mt * 8 + v] = ldsD[mt * 16 + rb + v];

  const __bf16* We = Wt + (size_t)e * H * H;
  const __bf16* ar = ldsA + (size_t)(lane & 15) * LSTR;
#pragma unroll
  for (int ct = 0; ct < 4; ct++) {
    int cb = ((wave << 2) + ct) << 4;
    const __bf16* brow = We + (size_t)(cb + (lane & 15)) * H;
    v8f cacc[4] = {{}, {}, {}, {}};
#pragma unroll
    for (int k0 = 0; k0 < H; k0 += 32) {
      v16bf b = frag_row(brow, k0, lane);
      cacc[0] = wmma_bf16(frag_row(ar,             k0, lane), b, cacc[0]);
      cacc[1] = wmma_bf16(frag_row(ar + 16 * LSTR, k0, lane), b, cacc[1]);
      cacc[2] = wmma_bf16(frag_row(ar + 32 * LSTR, k0, lane), b, cacc[2]);
      cacc[3] = wmma_bf16(frag_row(ar + 48 * LSTR, k0, lane), b, cacc[3]);
    }
    int col = cb + (lane & 15);
#pragma unroll
    for (int mt = 0; mt < 4; mt++)
#pragma unroll
      for (int v = 0; v < 8; v++) {
        int dn = dnreg[mt * 8 + v];
        if (dn >= 0) unsafeAtomicAdd(&a[(size_t)dn * H + col], cacc[mt][v]);
      }
  }
}

// ---- fused GRU: 32 nodes/WG; converts a on the fly; writes h (f32) + hb (bf16)
__global__ __launch_bounds__(128) void k_gru(
    __bf16* __restrict__ hb, const float* __restrict__ abuf,
    const __bf16* __restrict__ Wih, const __bf16* __restrict__ Whh,
    const float* __restrict__ bih, const float* __restrict__ bhh,
    float* __restrict__ h)
{
  __shared__ __bf16 lds[64 * LSTR];
  __bf16* ldsA = lds;                 // 32 rows of a (bf16)
  __bf16* ldsH = lds + 32 * LSTR;     // 32 rows of h (bf16)
  int nb  = blockIdx.x * 32;
  int tid = threadIdx.x;

  // stage a: convert f32 -> bf16 while copying (32 rows x 256)
  for (int i = tid; i < 2048; i += 128) {
    int row = i >> 6, c4 = (i & 63) << 2;
    int srow = nb + row; if (srow >= NN) srow = NN - 1;
    float4 v = *(const float4*)(abuf + (size_t)srow * H + c4);
    __bf16* d = ldsA + row * LSTR + c4;
    d[0] = (__bf16)v.x; d[1] = (__bf16)v.y; d[2] = (__bf16)v.z; d[3] = (__bf16)v.w;
  }
  // stage h (already bf16)
  for (int i = tid; i < 512; i += 128) {
    int row = i >> 4, ch = (i & 15) << 4;
    int srow = nb + row; if (srow >= NN) srow = NN - 1;
    const bf16x8* sp = (const bf16x8*)(hb + (size_t)srow * H + ch);
    bf16x8* dp = (bf16x8*)(ldsH + row * LSTR + ch);
    dp[0] = sp[0]; dp[1] = sp[1];
  }
  __syncthreads();

  int lane = tid & 31, wave = tid >> 5;
  const __bf16* ar = ldsA + (size_t)(lane & 15) * LSTR;
  const __bf16* hr = ldsH + (size_t)(lane & 15) * LSTR;
  for (int it = 0; it < 4; it++) {
    int cb = ((it << 2) + wave) << 4;
    int cc = cb + (lane & 15);
    const __bf16* br_r = Wih + (size_t)cc * H;
    const __bf16* br_z = Wih + (size_t)(cc + H) * H;
    const __bf16* br_n = Wih + (size_t)(cc + 2 * H) * H;
    const __bf16* bh_r = Whh + (size_t)cc * H;
    const __bf16* bh_z = Whh + (size_t)(cc + H) * H;
    const __bf16* bh_n = Whh + (size_t)(cc + 2 * H) * H;
    v8f cxr[2] = {{}, {}}, cxz[2] = {{}, {}}, cxn[2] = {{}, {}};
    v8f chrv[2] = {{}, {}}, chzv[2] = {{}, {}}, chnv[2] = {{}, {}};
#pragma unroll
    for (int k0 = 0; k0 < H; k0 += 32) {
      v16bf a0 = frag_row(ar,             k0, lane);
      v16bf a1 = frag_row(ar + 16 * LSTR, k0, lane);
      v16bf h0 = frag_row(hr,             k0, lane);
      v16bf h1 = frag_row(hr + 16 * LSTR, k0, lane);
      v16bf b;
      b = frag_row(br_r, k0, lane); cxr[0] = wmma_bf16(a0, b, cxr[0]); cxr[1] = wmma_bf16(a1, b, cxr[1]);
      b = frag_row(br_z, k0, lane); cxz[0] = wmma_bf16(a0, b, cxz[0]); cxz[1] = wmma_bf16(a1, b, cxz[1]);
      b = frag_row(br_n, k0, lane); cxn[0] = wmma_bf16(a0, b, cxn[0]); cxn[1] = wmma_bf16(a1, b, cxn[1]);
      b = frag_row(bh_r, k0, lane); chrv[0] = wmma_bf16(h0, b, chrv[0]); chrv[1] = wmma_bf16(h1, b, chrv[1]);
      b = frag_row(bh_z, k0, lane); chzv[0] = wmma_bf16(h0, b, chzv[0]); chzv[1] = wmma_bf16(h1, b, chzv[1]);
      b = frag_row(bh_n, k0, lane); chnv[0] = wmma_bf16(h0, b, chnv[0]); chnv[1] = wmma_bf16(h1, b, chnv[1]);
    }
    float bir = bih[cc], biz = bih[cc + H], bin_ = bih[cc + 2 * H];
    float bhr = bhh[cc], bhz = bhh[cc + H], bhn  = bhh[cc + 2 * H];
    int rb = (lane >> 4) << 3;
#pragma unroll
    for (int mt = 0; mt < 2; mt++)
#pragma unroll
      for (int v = 0; v < 8; v++) {
        int node = nb + mt * 16 + rb + v;
        if (node < NN) {
          float r   = sigm(cxr[mt][v] + bir + chrv[mt][v] + bhr);
          float z   = sigm(cxz[mt][v] + biz + chzv[mt][v] + bhz);
          float nn2 = tanhf(cxn[mt][v] + bin_ + r * (chnv[mt][v] + bhn));
          float* hp = h + (size_t)node * H + cc;
          float hold = *hp;
          float hnew = (1.0f - z) * nn2 + z * hold;
          *hp = hnew;
          hb[(size_t)node * H + cc] = (__bf16)hnew;
        }
      }
  }
}

// ---- readout ---------------------------------------------------------------
__global__ void k_relu_scatter(const float* __restrict__ h, const int* __restrict__ gid,
                               float* __restrict__ hg) {
  int idx = blockIdx.x * 256 + threadIdx.x;
  if (idx < NN * H) {
    int n = idx >> 8, c = idx & 255;
    float v = h[idx]; v = v > 0.0f ? v : 0.0f;
    unsafeAtomicAdd(&hg[(size_t)gid[n] * H + c], v);
  }
}

__global__ void k_out(const float* __restrict__ hg, const float* __restrict__ Wc,
                      const float* __restrict__ bc, float* __restrict__ out) {
  __shared__ float red[256];
  int g = blockIdx.x, t = threadIdx.x;
  red[t] = hg[(size_t)g * H + t] * Wc[t];
  __syncthreads();
  for (int s = 128; s; s >>= 1) { if (t < s) red[t] += red[t + s]; __syncthreads(); }
  if (t == 0) out[g] = 1.0f / (1.0f + __expf(-(red[0] + bc[0])));
}

// ---- launch ----------------------------------------------------------------
extern "C" void kernel_launch(void* const* d_in, const int* in_sizes, int n_in,
                              void* d_out, int out_size, void* d_ws, size_t ws_size,
                              hipStream_t stream) {
  const float* feat = (const float*)d_in[0];
  const float* W    = (const float*)d_in[1];
  const float* Wih  = (const float*)d_in[2];
  const float* Whh  = (const float*)d_in[3];
  const float* bih  = (const float*)d_in[4];
  const float* bhh  = (const float*)d_in[5];
  const float* Wc   = (const float*)d_in[6];
  const float* bc   = (const float*)d_in[7];
  const int* srcp   = (const int*)d_in[8];
  const int* dstp   = (const int*)d_in[9];
  const int* ety    = (const int*)d_in[10];
  const int* gid    = (const int*)d_in[11];
  float* out = (float*)d_out;

  char* p = (char*)d_ws;
  auto carve = [&](size_t bytes) {
    char* r = p; p += (bytes + 255) & ~(size_t)255; return r;
  };
  float*  hbuf = (float*)carve(sizeof(float) * (size_t)NN * H);
  float*  abuf = (float*)carve(sizeof(float) * (size_t)NN * H);
  __bf16* hb   = (__bf16*)carve(2ull * NN * H);
  __bf16* Wt   = (__bf16*)carve(2ull * NT * H * H);
  __bf16* WihB = (__bf16*)carve(2ull * 3 * H * H);
  __bf16* WhhB = (__bf16*)carve(2ull * 3 * H * H);
  int*    perm = (int*)carve(4ull * NE);
  int*    meta = (int*)carve(256);
  float*  hg   = (float*)carve(sizeof(float) * (size_t)NG * H);

  dim3 b256(256), b128(128);

  k_zero_meta<<<1, 64, 0, stream>>>(meta);
  k_hist<<<(NE + 255) / 256, b256, 0, stream>>>(ety, meta);
  k_scan<<<1, 1, 0, stream>>>(meta);
  k_scatter<<<(NE + 255) / 256, b256, 0, stream>>>(ety, meta, perm);
  k_conv_wt<<<(NT * H * H + 255) / 256, b256, 0, stream>>>(W, Wt);
  k_f2b<<<(3 * H * H + 1023) / 1024, b256, 0, stream>>>(Wih, WihB, 3 * H * H);
  k_f2b<<<(3 * H * H + 1023) / 1024, b256, 0, stream>>>(Whh, WhhB, 3 * H * H);
  k_inith<<<(NN * H + 255) / 256, b256, 0, stream>>>(feat, hbuf, hb);

  for (int s = 0; s < NSTEP; s++) {
    k_zerof<<<(NN * H + 1023) / 1024, b256, 0, stream>>>(abuf, NN * H);
    k_msg<<<MAXTILES, b128, 0, stream>>>(hb, Wt, perm, srcp, dstp, meta, abuf);
    k_gru<<<(NN + 31) / 32, b128, 0, stream>>>(hb, abuf, WihB, WhhB, bih, bhh, hbuf);
  }

  k_zerof<<<(NG * H + 1023) / 1024, b256, 0, stream>>>(hg, NG * H);
  k_relu_scatter<<<(NN * H + 255) / 256, b256, 0, stream>>>(hbuf, gid, hg);
  k_out<<<NG, b256, 0, stream>>>(hg, Wc, bc, out);
}
